// GraphAutoencoder_76819785056523
// MI455X (gfx1250) — matrix operand order repaired and checked by reading, not compiled, not run on hardware
//
#include <hip/hip_runtime.h>

typedef __attribute__((ext_vector_type(16))) _Float16 v16h;
typedef __attribute__((ext_vector_type(8)))  _Float16 v8h;
typedef __attribute__((ext_vector_type(8)))  float    v8f;

#define N_NODES 12000
#define N_EDGES 384000
#define IN_DIM  500
#define H1D     128
#define H2D     64
#define KPAD1   512   // IN_DIM padded to multiple of 32
#define MPAD4   512   // IN_DIM (output cols of W4) padded to multiple of 16

// ---------------------------------------------------------------- utilities
__global__ void fill_f32(float* __restrict__ p, float v, long n) {
  long i = (long)blockIdx.x * blockDim.x + threadIdx.x;
  if (i < n) p[i] = v;
}

__global__ void degree_accum(const int* __restrict__ ei, float* __restrict__ deg, int E) {
  int e = blockIdx.x * blockDim.x + threadIdx.x;
  if (e < E) atomicAdd(&deg[ei[E + e]], 1.0f);
}

__global__ void rsqrt_inplace(float* __restrict__ deg, int n) {
  int i = blockIdx.x * blockDim.x + threadIdx.x;
  if (i < n) {
    float d = deg[i];
    deg[i] = (d > 0.0f) ? rsqrtf(d) : 0.0f;
  }
}

// f32 -> f16 row-major copy with zero padding: dst[r*ldd+k], r<rowsPad, k<ldd
__global__ void cvt_f16_pad(const float* __restrict__ src, int lds,
                            _Float16* __restrict__ dst, int ldd,
                            long rowsPad, long rows, int K) {
  long i = (long)blockIdx.x * blockDim.x + threadIdx.x;
  if (i >= rowsPad * ldd) return;
  long r = i / ldd; int k = (int)(i % ldd);
  float v = (r < rows && k < K) ? src[r * lds + k] : 0.0f;
  dst[i] = (_Float16)v;
}

// W[K x M] row-major -> BT[m*ldd + k] = W[k*M+m], f16, zero padded to Mpad x ldd
__global__ void cvt_f16_transpose(const float* __restrict__ W, int K, int M,
                                  _Float16* __restrict__ dst, int ldd, int Mpad) {
  long i = (long)blockIdx.x * blockDim.x + threadIdx.x;
  if (i >= (long)Mpad * ldd) return;
  int m = (int)(i / ldd); int k = (int)(i % ldd);
  float v = (m < M && k < K) ? W[(long)k * M + m] : 0.0f;
  dst[i] = (_Float16)v;
}

// out[dst,:] += h[src,:] * dinv[src]*dinv[dst]; edges [0,E) real, [E,E+n) self loops
__global__ void scatter_msgs(const float* __restrict__ h, const int* __restrict__ ei,
                             const float* __restrict__ dinv, float* __restrict__ out,
                             int E, int n, int F4) {
  long tid = (long)blockIdx.x * blockDim.x + threadIdx.x;
  long total = (long)(E + n) * F4;
  if (tid >= total) return;
  int e  = (int)(tid / F4);
  int fc = (int)(tid % F4) * 4;
  int s, d;
  if (e < E) { s = ei[e]; d = ei[E + e]; } else { s = d = e - E; }
  float w = dinv[s] * dinv[d];
  int F = F4 * 4;
  float4 v = *(const float4*)(h + (long)s * F + fc);
  float* op = out + (long)d * F + fc;
  atomicAdd(op + 0, v.x * w);
  atomicAdd(op + 1, v.y * w);
  atomicAdd(op + 2, v.z * w);
  atomicAdd(op + 3, v.w * w);
}

// y = agg + b (f32, optional relu) -- used for x_hat final output
__global__ void bias_act(float* __restrict__ y, const float* __restrict__ b,
                         long rows, int F, int relu) {
  long i = (long)blockIdx.x * blockDim.x + threadIdx.x;
  if (i >= rows * (long)F) return;
  float v = y[i] + b[(int)(i % F)];
  if (relu) v = fmaxf(v, 0.0f);
  y[i] = v;
}

// out_h = f16(relu(agg + b)) -- activations only feed the next GEMM, so store f16
__global__ void bias_relu_f16(const float* __restrict__ agg, const float* __restrict__ b,
                              _Float16* __restrict__ out, long rows, int F) {
  long i = (long)blockIdx.x * blockDim.x + threadIdx.x;
  if (i >= rows * (long)F) return;
  out[i] = (_Float16)fmaxf(agg[i] + b[(int)(i % F)], 0.0f);
}

// ---------------------------------------------------------------- WMMA GEMM
// D[r,c] = sum_k A[r,k]*BT[c,k].  A: f16 row-major [rows x K], K % 32 == 0.
// BT: f16 [colsPad x K] (column c of B stored contiguously; zero padded).
// One 16x16 tile per wave. Per 32-K step: 4x global_load_b128 + 1 WMMA, no
// branches (EXEC all-1s). Store bounds-checked on cols only (rows % 16 == 0).
__global__ void wmma_gemm_f16(const _Float16* __restrict__ A, int lda,
                              const _Float16* __restrict__ BT, int ldbt,
                              float* __restrict__ D, long ldd,
                              int rows, int cols, int K, int tilesN, int ntStore) {
  int wave = threadIdx.x >> 5;
  long tile = (long)blockIdx.x * (blockDim.x >> 5) + wave;
  long nTiles = (long)(rows >> 4) * tilesN;
  if (tile >= nTiles) return;               // wave-uniform
  int tm = (int)(tile / tilesN);
  int tn = (int)(tile % tilesN);
  int lane = threadIdx.x & 31;
  int half = lane >> 4;
  int r    = lane & 15;
  const _Float16* aRow = A  + (long)(tm * 16 + r) * lda;   // A row per lane
  const _Float16* bCol = BT + (long)(tn * 16 + r) * ldbt;  // B column per lane
  v8f acc = {};
  for (int k0 = 0; k0 < K; k0 += 32) {
    // A 16-bit 16x32 layout: e=0..7 -> K = k0+8*half+(0..7); e=8..15 -> +16.
    v8h a_lo = *(const v8h*)(aRow + k0 + 8 * half);
    v8h a_hi = *(const v8h*)(aRow + k0 + 16 + 8 * half);
    v16h a = __builtin_shufflevector(a_lo, a_hi, 0, 1, 2, 3, 4, 5, 6, 7,
                                     8, 9, 10, 11, 12, 13, 14, 15);
    // B 16-bit 32x16 layout: e=0..7 -> K = k0+8*half+(0..7); e=8..15 -> +16.
    v8h b_lo = *(const v8h*)(bCol + k0 + 8 * half);
    v8h b_hi = *(const v8h*)(bCol + k0 + 16 + 8 * half);
    v16h b = __builtin_shufflevector(b_lo, b_hi, 0, 1, 2, 3, 4, 5, 6, 7,
                                     8, 9, 10, 11, 12, 13, 14, 15);
    acc = __builtin_amdgcn_wmma_f32_16x16x32_f16(
        false, a, false, b, (short)0, acc, false, false);
  }
  int col = tn * 16 + r;
  if (col < cols) {
#pragma unroll
    for (int j = 0; j < 8; ++j) {           // C/D: VGPR j -> row M = j + 8*half
      float* p = D + (long)(tm * 16 + j + 8 * half) * ldd + col;
      if (ntStore) __builtin_nontemporal_store(acc[j], p);
      else         *p = acc[j];
    }
  }
}

// ---------------------------------------------------------------- host side
static inline int cdiv(long a, long b) { return (int)((a + b - 1) / b); }

static void launch_gemm(const _Float16* A, int lda, const _Float16* BT, int ldbt,
                        float* D, long ldd, int rows, int cols, int K,
                        int ntStore, hipStream_t s) {
  int tilesN = (cols + 15) / 16;
  long tiles = (long)(rows / 16) * tilesN;
  int blocks = (int)((tiles + 7) / 8);      // 8 waves (16x16 tiles) per block
  wmma_gemm_f16<<<blocks, 256, 0, s>>>(A, lda, BT, ldbt, D, ldd, rows, cols, K,
                                       tilesN, ntStore);
}

extern "C" void kernel_launch(void* const* d_in, const int* in_sizes, int n_in,
                              void* d_out, int out_size, void* d_ws, size_t ws_size,
                              hipStream_t stream) {
  const float* x  = (const float*)d_in[0];
  const int*   ei = (const int*)d_in[1];     // edge_index [2, E]
  const float* W1 = (const float*)d_in[2];  const float* b1 = (const float*)d_in[3];
  const float* W2 = (const float*)d_in[4];  const float* b2 = (const float*)d_in[5];
  const float* W3 = (const float*)d_in[6];  const float* b3 = (const float*)d_in[7];
  const float* W4 = (const float*)d_in[8];  const float* b4 = (const float*)d_in[9];
  const float* W5 = (const float*)d_in[10]; const float* b5 = (const float*)d_in[11];

  float* out   = (float*)d_out;
  float* x_hat = out;                                  // [N, IN_DIM] f32
  float* a_hat = out + (long)N_NODES * IN_DIM;         // [N, N] f32

  // ---- workspace carve (256B aligned)
  char* base = (char*)d_ws;
  auto alloc = [&](size_t bytes) { char* p = base; base += (bytes + 255) & ~(size_t)255; return p; };
  float*    htmp = (float*)   alloc((size_t)N_NODES * 512 * 4);  // pre-agg GEMM out
  float*    agg  = (float*)   alloc((size_t)N_NODES * H1D * 4);  // aggregation buf
  float*    dinv = (float*)   alloc((size_t)N_NODES * 4);
  _Float16* x_h  = (_Float16*)alloc((size_t)N_NODES * KPAD1 * 2);
  _Float16* z1_h = (_Float16*)alloc((size_t)N_NODES * H1D * 2);
  _Float16* z2_h = (_Float16*)alloc((size_t)N_NODES * H2D * 2);
  _Float16* xh_h = (_Float16*)alloc((size_t)N_NODES * H1D * 2);
  _Float16* zs_h = (_Float16*)alloc((size_t)N_NODES * H2D * 2);
  _Float16* W1t  = (_Float16*)alloc((size_t)H1D * KPAD1 * 2);
  _Float16* W2t  = (_Float16*)alloc((size_t)H2D * H1D * 2);
  _Float16* W3t  = (_Float16*)alloc((size_t)H1D * H2D * 2);
  _Float16* W4t  = (_Float16*)alloc((size_t)MPAD4 * H1D * 2);
  _Float16* W5t  = (_Float16*)alloc((size_t)H2D * H2D * 2);

  const int  TB = 256;
  const long EN = (long)(N_EDGES + N_NODES);
  const long NN = (long)N_NODES;

  // ---- GCN normalization: deg (self loop = 1) -> dinv = rsqrt(deg)
  fill_f32<<<cdiv(NN, TB), TB, 0, stream>>>(dinv, 1.0f, NN);
  degree_accum<<<cdiv(N_EDGES, TB), TB, 0, stream>>>(ei, dinv, N_EDGES);
  rsqrt_inplace<<<cdiv(NN, TB), TB, 0, stream>>>(dinv, N_NODES);

  // ---- f16 operand prep (x padded K 500->512; weights transposed + padded)
  cvt_f16_pad<<<cdiv(NN * KPAD1, TB), TB, 0, stream>>>(x, IN_DIM, x_h, KPAD1, NN, NN, IN_DIM);
  cvt_f16_transpose<<<cdiv((long)H1D * KPAD1, TB), TB, 0, stream>>>(W1, IN_DIM, H1D, W1t, KPAD1, H1D);
  cvt_f16_transpose<<<cdiv((long)H2D * H1D, TB), TB, 0, stream>>>(W2, H1D, H2D, W2t, H1D, H2D);
  cvt_f16_transpose<<<cdiv((long)H1D * H2D, TB), TB, 0, stream>>>(W3, H2D, H1D, W3t, H2D, H1D);
  cvt_f16_transpose<<<cdiv((long)MPAD4 * H1D, TB), TB, 0, stream>>>(W4, H1D, IN_DIM, W4t, H1D, MPAD4);
  cvt_f16_transpose<<<cdiv((long)H2D * H2D, TB), TB, 0, stream>>>(W5, H2D, H2D, W5t, H2D, H2D);

  // ---- conv1: z1 = relu(agg(x @ W1) + b1)
  launch_gemm(x_h, KPAD1, W1t, KPAD1, htmp, H1D, N_NODES, H1D, KPAD1, 0, stream);
  fill_f32<<<cdiv(NN * H1D, TB), TB, 0, stream>>>(agg, 0.0f, NN * H1D);
  scatter_msgs<<<cdiv(EN * (H1D / 4), TB), TB, 0, stream>>>(htmp, ei, dinv, agg, N_EDGES, N_NODES, H1D / 4);
  bias_relu_f16<<<cdiv(NN * H1D, TB), TB, 0, stream>>>(agg, b1, z1_h, NN, H1D);

  // ---- conv2: z2 = relu(agg(z1 @ W2) + b2)
  launch_gemm(z1_h, H1D, W2t, H1D, htmp, H2D, N_NODES, H2D, H1D, 0, stream);
  fill_f32<<<cdiv(NN * H2D, TB), TB, 0, stream>>>(agg, 0.0f, NN * H2D);
  scatter_msgs<<<cdiv(EN * (H2D / 4), TB), TB, 0, stream>>>(htmp, ei, dinv, agg, N_EDGES, N_NODES, H2D / 4);
  bias_relu_f16<<<cdiv(NN * H2D, TB), TB, 0, stream>>>(agg, b2, z2_h, NN, H2D);

  // ---- conv3: xh = relu(agg(z2 @ W3) + b3)
  launch_gemm(z2_h, H2D, W3t, H2D, htmp, H1D, N_NODES, H1D, H2D, 0, stream);
  fill_f32<<<cdiv(NN * H1D, TB), TB, 0, stream>>>(agg, 0.0f, NN * H1D);
  scatter_msgs<<<cdiv(EN * (H1D / 4), TB), TB, 0, stream>>>(htmp, ei, dinv, agg, N_EDGES, N_NODES, H1D / 4);
  bias_relu_f16<<<cdiv(NN * H1D, TB), TB, 0, stream>>>(agg, b3, xh_h, NN, H1D);

  // ---- conv4: x_hat = agg(xh @ W4) + b4  (aggregates directly into d_out, f32)
  launch_gemm(xh_h, H1D, W4t, H1D, htmp, IN_DIM, N_NODES, IN_DIM, H1D, 0, stream);
  fill_f32<<<cdiv(NN * IN_DIM, TB), TB, 0, stream>>>(x_hat, 0.0f, NN * IN_DIM);
  scatter_msgs<<<cdiv(EN * (IN_DIM / 4), TB), TB, 0, stream>>>(htmp, ei, dinv, x_hat, N_EDGES, N_NODES, IN_DIM / 4);
  bias_act<<<cdiv(NN * IN_DIM, TB), TB, 0, stream>>>(x_hat, b4, NN, IN_DIM, 0);

  // ---- conv5: zs = relu(agg(z2 @ W5) + b5)
  launch_gemm(z2_h, H2D, W5t, H2D, htmp, H2D, N_NODES, H2D, H2D, 0, stream);
  fill_f32<<<cdiv(NN * H2D, TB), TB, 0, stream>>>(agg, 0.0f, NN * H2D);
  scatter_msgs<<<cdiv(EN * (H2D / 4), TB), TB, 0, stream>>>(htmp, ei, dinv, agg, N_EDGES, N_NODES, H2D / 4);
  bias_relu_f16<<<cdiv(NN * H2D, TB), TB, 0, stream>>>(agg, b5, zs_h, NN, H2D);

  // ---- a_hat = zs @ zs.T : zs_h is simultaneously A (row-major) and BT
  // (transposed access == its natural layout). 576 MB streaming write -> NT stores.
  launch_gemm(zs_h, H2D, zs_h, H2D, a_hat, N_NODES, N_NODES, N_NODES, H2D, 1, stream);
}
